// NNFOwithBayesianJumps_20023137534283
// MI455X (gfx1250) — matrix operand order repaired and checked by reading, not compiled, not run on hardware
//
#include <hip/hip_runtime.h>

// ---------------- model constants ----------------
#define B_      2048
#define H_      768
#define D_      96
#define TWOD    192
#define PH_     512
#define PREP_   10
#define PREPD   960     // PREP_*D_
#define COV_    32
#define COVH_   64
#define NT_     50
#define OBS_    1024
#define STEPS_  2
#define DT_     0.05f
#define LOGC_   0.91893853320467274178f   // log(sqrt(2*pi))
#define LOGNS_  -4.6051701859880913680f   // log(0.01)
#define INV2N2  5000.0f                   // 1/(2*0.01^2)

#define BH      (B_*H_)        // 1572864
#define B2D     (B_*TWOD)      // 393216

typedef __bf16 bf16_t;
typedef __bf16 v8bf  __attribute__((ext_vector_type(8)));
typedef __bf16 v16bf __attribute__((ext_vector_type(16)));
typedef float  v8f   __attribute__((ext_vector_type(8)));

__device__ __forceinline__ float sigm_(float x) { return 1.0f / (1.0f + __expf(-x)); }

// ---- 16-byte global->LDS copy: async (ASYNCcnt) if available, else sync ----
#if __has_builtin(__builtin_amdgcn_global_load_async_to_lds_b128)
#define USE_ASYNC_LDS 1
// Builtin prototype (from clang diagnostic): (v4i as1*, v4i as3*, Imm, Imm)
typedef int v4i_ __attribute__((vector_size(16)));
typedef __attribute__((address_space(1))) v4i_ gv4i;
typedef __attribute__((address_space(3))) v4i_ lv4i;
#else
#define USE_ASYNC_LDS 0
#endif

__device__ __forceinline__ void cp16_g2l(const bf16_t* __restrict__ g, bf16_t* l)
{
#if USE_ASYNC_LDS
    __builtin_amdgcn_global_load_async_to_lds_b128((gv4i*)g, (lv4i*)l, 0, 0);
#else
    *(v8bf*)l = *(const v8bf*)g;
#endif
}

__device__ __forceinline__ void cp16_wait()
{
#if USE_ASYNC_LDS
#if __has_builtin(__builtin_amdgcn_s_wait_asynccnt)
    __builtin_amdgcn_s_wait_asynccnt(0);
#else
    asm volatile("s_wait_asynccnt 0x0" ::: "memory");
#endif
#endif
}

// =====================================================================
// WMMA GEMM: act( A[M,K] @ B[K,N] + bias + (ACCUM ? C_old : 0) )
//   A : bf16 row-major [M,K]  (activation shadow, produced by epilogues)
//   Bt: bf16 transposed [N,K] (weights, pre-converted once per call)
//   C : optional fp32 out; Cb: optional bf16 out (feeds next GEMM)
// Block tile 128(M) x 64(N), K-step 32; 4 waves in 2x2, wave tile 64x32 =
// 4x2 WMMA tiles. Double-buffered LDS, async fill, 1 barrier per K-step.
// Requires M%128==0, N%64==0, K%32==0 (true for all shapes here).
// ACT: 0 none, 1 relu, 2 tanh, 3 sigmoid
// =====================================================================
template <int ACT, bool ACCUM>
__global__ __launch_bounds__(128)
void wmma_gemm(const bf16_t* __restrict__ A, const bf16_t* __restrict__ Bt,
               float* __restrict__ C, bf16_t* __restrict__ Cb,
               const float* __restrict__ bias, int M, int N, int K)
{
    __shared__ bf16_t As [2][128][40];   // [buf][row][k]  (80B rows, 16B aligned)
    __shared__ bf16_t BsT[2][ 64][40];   // [buf][col][k]

    const int tid  = threadIdx.x;
    const int lane = tid & 31;
    const int wave = tid >> 5;
    const int wm   = (wave >> 1) * 64;
    const int wn   = (wave & 1)  * 32;

    const int bm = blockIdx.y * 128;
    const int bn = blockIdx.x * 64;

    v8f acc[4][2];
    #pragma unroll
    for (int i = 0; i < 4; ++i)
        #pragma unroll
        for (int j = 0; j < 2; ++j)
            #pragma unroll
            for (int e = 0; e < 8; ++e) acc[i][j][e] = 0.0f;

    // Stage one 128x32 A tile + 64x32 B tile (16B chunks, 6 per thread).
    auto stage = [&](int sb, int k0) {
        #pragma unroll
        for (int it = 0; it < 4; ++it) {           // A: 512 chunks
            const int q = it * 128 + tid;
            const int r = q >> 2, o = (q & 3) << 3;
            cp16_g2l(A + (size_t)(bm + r) * K + k0 + o, &As[sb][r][o]);
        }
        #pragma unroll
        for (int it = 0; it < 2; ++it) {           // B: 256 chunks
            const int q = it * 128 + tid;
            const int c = q >> 2, o = (q & 3) << 3;
            cp16_g2l(Bt + (size_t)(bn + c) * K + k0 + o, &BsT[sb][c][o]);
        }
    };

    stage(0, 0);
    int cur = 0;
    for (int k0 = 0; k0 < K; k0 += 32) {
        cp16_wait();          // my tile-fill complete (no-op in sync mode)
        __syncthreads();      // everyone's fill complete, prior reads done
        if (k0 + 32 < K) stage(cur ^ 1, k0 + 32);   // overlap next fill

        // A fragment (16-bit A 16x32, wave32):
        //   lanes 0-15: M=lane,    elems 0..7 -> K 0..7,  8..15 -> K 16..23
        //   lanes16-31: M=lane-16, elems 0..7 -> K 8..15, 8..15 -> K 24..31
        const int am   = lane & 15;
        const int koff = (lane >> 4) << 3;
        v16bf af[4];
        #pragma unroll
        for (int t = 0; t < 4; ++t) {
            const bf16_t* base = &As[cur][wm + t * 16 + am][0];
            v8bf lo = *(const v8bf*)(base + koff);
            v8bf hi = *(const v8bf*)(base + koff + 16);
            af[t] = __builtin_shufflevector(lo, hi,
                     0,1,2,3,4,5,6,7,8,9,10,11,12,13,14,15);
        }
        // B fragment (32x16): same K order as A (lanes 0-15 K0..7,16..23; lanes 16-31 K8..15,24..31).
        const int bnn = lane & 15;
        const int kh  = (lane >> 4) << 3;
        v16bf bfr[2];
        #pragma unroll
        for (int t = 0; t < 2; ++t) {
            const bf16_t* base = &BsT[cur][wn + t * 16 + bnn][0] + kh;
            v8bf lo = *(const v8bf*)(base);
            v8bf hi = *(const v8bf*)(base + 16);
            bfr[t] = __builtin_shufflevector(lo, hi,
                      0,1,2,3,4,5,6,7,8,9,10,11,12,13,14,15);
        }

        #pragma unroll
        for (int i = 0; i < 4; ++i)
            #pragma unroll
            for (int j = 0; j < 2; ++j)
                acc[i][j] = __builtin_amdgcn_wmma_f32_16x16x32_bf16(
                    false, af[i], false, bfr[j],
                    (short)0, acc[i][j], false, false);
        cur ^= 1;
    }

    // Epilogue. f32 C/D layout: row = (lane>>4)*8 + r, col = lane&15.
    const int cn = lane & 15;
    const int mb = (lane >> 4) << 3;
    #pragma unroll
    for (int i = 0; i < 4; ++i)
        #pragma unroll
        for (int j = 0; j < 2; ++j) {
            const int col = bn + wn + j * 16 + cn;
            #pragma unroll
            for (int r = 0; r < 8; ++r) {
                const int row = bm + wm + i * 16 + mb + r;
                const size_t idx = (size_t)row * N + col;
                float v = acc[i][j][r];
                if (ACCUM) v += C[idx];
                if (bias)  v += bias[col];
                if (ACT == 1) v = fmaxf(v, 0.0f);
                else if (ACT == 2) v = tanhf(v);
                else if (ACT == 3) v = sigm_(v);
                if (C)  C[idx]  = v;
                if (Cb) Cb[idx] = (bf16_t)v;
            }
        }
}

// ---------------- weight prep: fp32 [K,N] -> bf16 transposed [N,K] -------
__global__ void k_w2bft(const float* __restrict__ W, bf16_t* __restrict__ Wt,
                        int K, int N)
{
    int i = blockIdx.x * blockDim.x + threadIdx.x;
    if (i < K * N) {
        int n = i / K, k = i - n * K;   // consecutive i -> consecutive k writes
        Wt[(size_t)n * K + k] = (bf16_t)W[(size_t)k * N + n];
    }
}

// fp32 -> bf16, same layout (for the cov input)
__global__ void k_f2bf(const float* __restrict__ a, bf16_t* __restrict__ o, int n)
{
    int i = blockIdx.x * blockDim.x + threadIdx.x;
    if (i < n) o[i] = (bf16_t)a[i];
}

// ---------------- elementwise / support kernels ----------------
__global__ void k_zero2(float* a) { a[0] = 0.0f; a[1] = 0.0f; }

// zh_bf = bf16(z * h)
__global__ void k_mul_bf(const float* __restrict__ z, const float* __restrict__ h,
                         bf16_t* __restrict__ zb, int n)
{
    int i = blockIdx.x * blockDim.x + threadIdx.x;
    if (i < n) zb[i] = (bf16_t)(z[i] * h[i]);
}

// h += DT*(1-z)*(n-h); also refresh bf16 shadow
__global__ void k_euler(float* __restrict__ h, bf16_t* __restrict__ hb,
                        const float* __restrict__ z, const float* __restrict__ n,
                        int nn)
{
    int i = blockIdx.x * blockDim.x + threadIdx.x;
    if (i < nn) {
        float hv = h[i], zv = z[i], nv = n[i];
        float o = hv + DT_ * (1.0f - zv) * (nv - hv);
        h[i]  = o;
        hb[i] = (bf16_t)o;
    }
}

// hobs_bf = bf16(h[idx])
__global__ void k_gather_bf(const float* __restrict__ h, const int* __restrict__ idx,
                            bf16_t* __restrict__ hobs)
{
    int i = blockIdx.x * blockDim.x + threadIdx.x;
    if (i < OBS_ * H_) {
        int r = i / H_, c = i - r * H_;
        hobs[i] = (bf16_t)h[(size_t)idx[r] * H_ + c];
    }
}

// Per (n,d): log-lik loss + feature prep  gi = bf16(relu(feats.w_prep+b)*mask)
__global__ void k_feats(const float* __restrict__ X, const float* __restrict__ Mt,
                        const int* __restrict__ idx, const float* __restrict__ p,
                        const float* __restrict__ w_prep, const float* __restrict__ bias_prep,
                        bf16_t* __restrict__ gi, float* __restrict__ l1)
{
    __shared__ float ssum;
    if (threadIdx.x == 0) ssum = 0.0f;
    __syncthreads();

    int i = blockIdx.x * blockDim.x + threadIdx.x;
    float loc = 0.0f;
    if (i < OBS_ * D_) {
        int n = i / D_, d = i - n * D_;
        int row = idx[n];
        float mean   = p[(size_t)row * TWOD + d];
        float logvar = p[(size_t)row * TWOD + D_ + d];
        float x = X[i], m = Mt[i];
        float sigma = __expf(0.5f * logvar);
        float err = (x - mean) / sigma;
        loc = 0.5f * ((err * err + logvar + 2.0f * LOGC_) * m);
        const float* w  = w_prep + (size_t)d * 4 * PREP_;
        const float* bp = bias_prep + (size_t)d * PREP_;
        bf16_t* go = gi + (size_t)n * PREPD + (size_t)d * PREP_;
        #pragma unroll
        for (int q = 0; q < PREP_; ++q) {
            float g = x * w[0 * PREP_ + q] + mean * w[1 * PREP_ + q]
                    + logvar * w[2 * PREP_ + q] + err * w[3 * PREP_ + q] + bp[q];
            go[q] = (bf16_t)(fmaxf(g, 0.0f) * m);
        }
    }
    atomicAdd(&ssum, loc);
    __syncthreads();
    if (threadIdx.x == 0) atomicAdd(l1, ssum);
}

// torch GRUCell combine + scatter into h (indices unique; reads old h at the
// scatter address before writing it — one thread per element, no hazard)
__global__ void k_combine(const float* __restrict__ gig, const float* __restrict__ gh,
                          const int* __restrict__ idx,
                          float* __restrict__ h, bf16_t* __restrict__ hb)
{
    int i = blockIdx.x * blockDim.x + threadIdx.x;
    if (i >= OBS_ * H_) return;
    int n = i / H_, c = i - n * H_;
    const size_t b3 = (size_t)n * 3 * H_;
    float ir = gig[b3 + c], iz = gig[b3 + H_ + c], in = gig[b3 + 2 * H_ + c];
    float hr = gh[b3 + c],  hz = gh[b3 + H_ + c],  hn = gh[b3 + 2 * H_ + c];
    float r  = sigm_(ir + hr);
    float z  = sigm_(iz + hz);
    float nn = tanhf(in + r * hn);
    const size_t ha = (size_t)idx[n] * H_ + c;
    float ho = h[ha];
    float o  = (1.0f - z) * nn + z * ho;
    h[ha]  = o;
    hb[ha] = (bf16_t)o;
}

__global__ void k_kl(const float* __restrict__ X, const float* __restrict__ Mt,
                     const int* __restrict__ idx, const float* __restrict__ p,
                     float* __restrict__ l2)
{
    __shared__ float ssum;
    if (threadIdx.x == 0) ssum = 0.0f;
    __syncthreads();

    int i = blockIdx.x * blockDim.x + threadIdx.x;
    float loc = 0.0f;
    if (i < OBS_ * D_) {
        int n = i / D_, d = i - n * D_;
        int row = idx[n];
        float mean2   = p[(size_t)row * TWOD + d];
        float logvar2 = p[(size_t)row * TWOD + D_ + d];
        float x = X[i], m = Mt[i];
        float var2 = __expf(logvar2);
        float dm = mean2 - x;
        float kl = LOGNS_ - 0.5f * logvar2 + (var2 + dm * dm) * INV2N2 - 0.5f;
        loc = kl * m;
    }
    atomicAdd(&ssum, loc);
    __syncthreads();
    if (threadIdx.x == 0) atomicAdd(l2, ssum);
}

// out = [ h (B*H) | loss | p (B*2D) | l1 ]
__global__ void k_emit(const float* __restrict__ h, const float* __restrict__ p,
                       const float* __restrict__ loss2, float* __restrict__ out)
{
    size_t i = (size_t)blockIdx.x * blockDim.x + threadIdx.x;
    if (i < (size_t)BH) out[i] = h[i];
    else if (i < (size_t)BH + B2D) out[(size_t)BH + 1 + (i - BH)] = p[i - BH];
    if (i == 0) {
        out[BH] = loss2[0] + loss2[1];
        out[(size_t)BH + 1 + B2D] = loss2[0];
    }
}

// ---------------- host-side launcher ----------------
static void launch_gemm(const bf16_t* A, const bf16_t* Bt, float* C, bf16_t* Cb,
                        const float* bias, int M, int N, int K, int act, bool accum,
                        hipStream_t s)
{
    dim3 g(N / 64, M / 128), b(128);
    if (!accum) {
        switch (act) {
            case 0: wmma_gemm<0, false><<<g, b, 0, s>>>(A, Bt, C, Cb, bias, M, N, K); break;
            case 1: wmma_gemm<1, false><<<g, b, 0, s>>>(A, Bt, C, Cb, bias, M, N, K); break;
            case 2: wmma_gemm<2, false><<<g, b, 0, s>>>(A, Bt, C, Cb, bias, M, N, K); break;
            default: wmma_gemm<3, false><<<g, b, 0, s>>>(A, Bt, C, Cb, bias, M, N, K); break;
        }
    } else {
        switch (act) {
            case 0: wmma_gemm<0, true><<<g, b, 0, s>>>(A, Bt, C, Cb, bias, M, N, K); break;
            case 1: wmma_gemm<1, true><<<g, b, 0, s>>>(A, Bt, C, Cb, bias, M, N, K); break;
            case 2: wmma_gemm<2, true><<<g, b, 0, s>>>(A, Bt, C, Cb, bias, M, N, K); break;
            default: wmma_gemm<3, true><<<g, b, 0, s>>>(A, Bt, C, Cb, bias, M, N, K); break;
        }
    }
}

static void launch_w2bft(const float* W, bf16_t* Wt, int K, int N, hipStream_t s)
{
    int n = K * N;
    k_w2bft<<<(n + 255) / 256, 256, 0, s>>>(W, Wt, K, N);
}

extern "C" void kernel_launch(void* const* d_in, const int* in_sizes, int n_in,
                              void* d_out, int out_size, void* d_ws, size_t ws_size,
                              hipStream_t stream)
{
    (void)in_sizes; (void)n_in; (void)out_size; (void)ws_size;

    const float* cov      = (const float*)d_in[0];
    const float* X        = (const float*)d_in[1];
    const float* Mm       = (const float*)d_in[2];
    const int*   obs_idx  = (const int*)  d_in[3];
    const float* cov_W1   = (const float*)d_in[4];
    const float* cov_b1   = (const float*)d_in[5];
    const float* cov_W2   = (const float*)d_in[6];
    const float* cov_b2   = (const float*)d_in[7];
    const float* pm_W1    = (const float*)d_in[8];
    const float* pm_b1    = (const float*)d_in[9];
    const float* pm_W2    = (const float*)d_in[10];
    const float* pm_b2    = (const float*)d_in[11];
    const float* gc_Wxz   = (const float*)d_in[12];
    const float* gc_bxz   = (const float*)d_in[13];
    const float* gc_Whz   = (const float*)d_in[14];
    const float* gc_Wxn   = (const float*)d_in[15];
    const float* gc_bxn   = (const float*)d_in[16];
    const float* gc_Whn   = (const float*)d_in[17];
    const float* go_Wih   = (const float*)d_in[18];
    const float* go_bih   = (const float*)d_in[19];
    const float* go_Whh   = (const float*)d_in[20];
    const float* go_bhh   = (const float*)d_in[21];
    const float* w_prep   = (const float*)d_in[22];
    const float* bias_prep= (const float*)d_in[23];

    // ---- workspace: fp32 region, then bf16 region (16B aligned) ----
    float* ws    = (float*)d_ws;
    float* h     = ws;                 ws += BH;
    float* p     = ws;                 ws += B2D;
    float* tz    = ws;                 ws += BH;                 // z gate
    float* tn    = ws;                 ws += BH;                 // n gate
    float* gig   = ws;                 ws += (size_t)OBS_ * 3 * H_;
    float* gh    = ws;                 ws += (size_t)OBS_ * 3 * H_;
    float* loss2 = ws;                 ws += 4;                  // [l1, l2, pad, pad]

    bf16_t* wb      = (bf16_t*)ws;     // all float sizes above are %4==0
    bf16_t* cov_bf  = wb;              wb += (size_t)B_ * COV_;
    bf16_t* h_bf    = wb;              wb += BH;
    bf16_t* p_bf    = wb;              wb += B2D;
    bf16_t* t1_bf   = wb;              wb += (size_t)B_ * PH_;
    bf16_t* zh_bf   = wb;              wb += BH;
    bf16_t* gi_bf   = wb;              wb += (size_t)OBS_ * PREPD;
    bf16_t* ho_bf   = wb;              wb += (size_t)OBS_ * H_;
    bf16_t* bt_c1   = wb;              wb += (size_t)COV_  * COVH_;
    bf16_t* bt_c2   = wb;              wb += (size_t)COVH_ * H_;
    bf16_t* bt_p1   = wb;              wb += (size_t)H_    * PH_;
    bf16_t* bt_p2   = wb;              wb += (size_t)PH_   * TWOD;
    bf16_t* bt_xz   = wb;              wb += (size_t)TWOD  * H_;
    bf16_t* bt_hz   = wb;              wb += (size_t)H_    * H_;
    bf16_t* bt_xn   = wb;              wb += (size_t)TWOD  * H_;
    bf16_t* bt_hn   = wb;              wb += (size_t)H_    * H_;
    bf16_t* bt_ih   = wb;              wb += (size_t)PREPD * 3 * H_;
    bf16_t* bt_hh   = wb;              wb += (size_t)H_    * 3 * H_;

    const int TB  = 256;
    const int gBH = (BH + TB - 1) / TB;
    const int gOH = (OBS_ * H_ + TB - 1) / TB;
    const int gOD = (OBS_ * D_ + TB - 1) / TB;

    // ---- once per call: weights -> bf16 [N][K]; cov -> bf16 ----
    launch_w2bft(cov_W1, bt_c1, COV_,  COVH_,   stream);
    launch_w2bft(cov_W2, bt_c2, COVH_, H_,      stream);
    launch_w2bft(pm_W1,  bt_p1, H_,    PH_,     stream);
    launch_w2bft(pm_W2,  bt_p2, PH_,   TWOD,    stream);
    launch_w2bft(gc_Wxz, bt_xz, TWOD,  H_,      stream);
    launch_w2bft(gc_Whz, bt_hz, H_,    H_,      stream);
    launch_w2bft(gc_Wxn, bt_xn, TWOD,  H_,      stream);
    launch_w2bft(gc_Whn, bt_hn, H_,    H_,      stream);
    launch_w2bft(go_Wih, bt_ih, PREPD, 3 * H_,  stream);
    launch_w2bft(go_Whh, bt_hh, H_,    3 * H_,  stream);
    k_f2bf<<<(B_ * COV_ + TB - 1) / TB, TB, 0, stream>>>(cov, cov_bf, B_ * COV_);

    k_zero2<<<1, 1, 0, stream>>>(loss2);

    // h0 = tanh(relu(cov @ cov_W1 + b1) @ cov_W2 + b2)
    launch_gemm(cov_bf, bt_c1, nullptr, t1_bf, cov_b1, B_, COVH_, COV_, 1, false, stream);
    launch_gemm(t1_bf, bt_c2, h, h_bf, cov_b2, B_, H_, COVH_, 2, false, stream);
    // p0 = p_model(h0)
    launch_gemm(h_bf, bt_p1, nullptr, t1_bf, pm_b1, B_, PH_, H_, 1, false, stream);
    launch_gemm(t1_bf, bt_p2, p, p_bf, pm_b2, B_, TWOD, PH_, 0, false, stream);

    for (int t = 0; t < NT_; ++t) {
        // ---- STEPS_ Euler ODE steps ----
        for (int s = 0; s < STEPS_; ++s) {
            // z = sigmoid(p@Wxz + bxz + h@Whz)
            launch_gemm(p_bf, bt_xz, tz, nullptr, gc_bxz, B_, H_, TWOD, 0, false, stream);
            launch_gemm(h_bf, bt_hz, tz, nullptr, nullptr, B_, H_, H_, 3, true, stream);
            // zh = z * h
            k_mul_bf<<<gBH, TB, 0, stream>>>(tz, h, zh_bf, BH);
            // n = tanh(p@Wxn + bxn + (z*h)@Whn)
            launch_gemm(p_bf, bt_xn, tn, nullptr, gc_bxn, B_, H_, TWOD, 0, false, stream);
            launch_gemm(zh_bf, bt_hn, tn, nullptr, nullptr, B_, H_, H_, 2, true, stream);
            // h += DT*(1-z)*(n-h)
            k_euler<<<gBH, TB, 0, stream>>>(h, h_bf, tz, tn, BH);
            // p = p_model(h)
            launch_gemm(h_bf, bt_p1, nullptr, t1_bf, pm_b1, B_, PH_, H_, 1, false, stream);
            launch_gemm(t1_bf, bt_p2, p, p_bf, pm_b2, B_, TWOD, PH_, 0, false, stream);
        }

        // ---- Bayesian observation update ----
        const float* Xt = X  + (size_t)t * OBS_ * D_;
        const float* Mt = Mm + (size_t)t * OBS_ * D_;
        const int*   it = obs_idx + (size_t)t * OBS_;

        k_feats<<<gOD, TB, 0, stream>>>(Xt, Mt, it, p, w_prep, bias_prep, gi_bf, loss2);
        k_gather_bf<<<gOH, TB, 0, stream>>>(h, it, ho_bf);
        launch_gemm(gi_bf, bt_ih, gig, nullptr, go_bih, OBS_, 3 * H_, PREPD, 0, false, stream);
        launch_gemm(ho_bf, bt_hh, gh,  nullptr, go_bhh, OBS_, 3 * H_, H_,    0, false, stream);
        k_combine<<<gOH, TB, 0, stream>>>(gig, gh, it, h, h_bf);
        // p = p_model(h)
        launch_gemm(h_bf, bt_p1, nullptr, t1_bf, pm_b1, B_, PH_, H_, 1, false, stream);
        launch_gemm(t1_bf, bt_p2, p, p_bf, pm_b2, B_, TWOD, PH_, 0, false, stream);
        // KL vs observation noise
        k_kl<<<gOD, TB, 0, stream>>>(Xt, Mt, it, p, loss2 + 1);
    }

    // ---- emit outputs: [h | loss | p | l1] ----
    const size_t totalCopy = (size_t)BH + B2D;
    k_emit<<<(int)((totalCopy + TB - 1) / TB), TB, 0, stream>>>(h, p, loss2, (float*)d_out);
}
